// FocusedDecoder_7335804142023
// MI455X (gfx1250) — matrix operand order repaired and checked by reading, not compiled, not run on hardware
//
#include <hip/hip_runtime.h>

typedef __bf16 bf16_t;
typedef __attribute__((ext_vector_type(16))) __bf16 v16bf;
typedef __attribute__((ext_vector_type(8)))  float  v8f;
typedef int v4i_vec __attribute__((vector_size(16)));

#define D_MODEL 256
#define NHEAD   8
#define HDIM    32
#define NQ      540
#define NQP     544        // 540 padded to 34*16
#define SRC_N   12800
#define DFF     1024
#define LAY     6

#define AS_GLOBAL __attribute__((address_space(1)))
#define AS_LDS    __attribute__((address_space(3)))

#if defined(__gfx1250__) && __has_builtin(__builtin_amdgcn_global_load_async_to_lds_b128) && \
    __has_builtin(__builtin_amdgcn_s_wait_asynccnt)
#define HAVE_ASYNC 1
#else
#define HAVE_ASYNC 0
#endif

__device__ __forceinline__ float neg_inf() { return -__builtin_inff(); }

// ---------------------------------------------------------------------------
// elementwise prep kernels
// ---------------------------------------------------------------------------
__global__ __launch_bounds__(256) void f32_to_bf16_k(const float* __restrict__ x,
                                                     bf16_t* __restrict__ y, int n) {
  int i = blockIdx.x * 256 + threadIdx.x;
  if (i < n) y[i] = (bf16_t)x[i];
}

// src/pos are [D][S] (channel-major); produce token-major bf16 s and s+pos
__global__ __launch_bounds__(256) void prep_src_k(const float* __restrict__ src,
                                                  const float* __restrict__ pos,
                                                  bf16_t* __restrict__ s_bf,
                                                  bf16_t* __restrict__ kp_bf) {
  int i = blockIdx.x * 256 + threadIdx.x;
  if (i >= SRC_N * D_MODEL) return;
  int t = i / D_MODEL, d = i % D_MODEL;
  float sv = src[(size_t)d * SRC_N + t];
  float pv = pos[(size_t)d * SRC_N + t];
  s_bf[i]  = (bf16_t)sv;
  kp_bf[i] = (bf16_t)(sv + pv);
}

// query_embed [NQ][2D] -> qpos (first D), tgt (second D); pad rows = 0
__global__ __launch_bounds__(256) void prep_q_k(const float* __restrict__ qe,
                                                float* __restrict__ tgt,
                                                float* __restrict__ qpos,
                                                bf16_t* __restrict__ tgt_bf,
                                                bf16_t* __restrict__ q_bf) {
  int i = blockIdx.x * 256 + threadIdx.x;
  if (i >= NQP * D_MODEL) return;
  int q = i / D_MODEL, d = i % D_MODEL;
  float t = 0.f, p = 0.f;
  if (q < NQ) {
    p = qe[(size_t)q * (2 * D_MODEL) + d];
    t = qe[(size_t)q * (2 * D_MODEL) + D_MODEL + d];
  }
  tgt[i] = t; qpos[i] = p;
  tgt_bf[i] = (bf16_t)t;
  q_bf[i]   = (bf16_t)(t + p);
}

__global__ __launch_bounds__(256) void copy_out_k(const float* __restrict__ tgt,
                                                  float* __restrict__ out, int n) {
  int i = blockIdx.x * 256 + threadIdx.x;
  if (i < n) out[i] = tgt[i];
}

// ---------------------------------------------------------------------------
// WMMA GEMM: C[M][N] = (A[M][K] @ W[N][K]^T + bias) * scale, optional relu.
// Block = 128 threads = 4 waves. Block tile = 64(M) x 64(N); each wave owns a
// 16x64 strip: one A fragment feeds 4 WMMAs. The shared 64x32 W tile is
// double-buffered in LDS via async global->LDS copies (ASYNCcnt) when the
// toolchain exposes the gfx1250 builtins; otherwise staged synchronously.
// Output fp32 (Cf) and/or bf16 (Cb); bf16 out may be stored transposed
// (Cb[col*ldOut + row]) to build V^T for attention.
// ---------------------------------------------------------------------------
__global__ __launch_bounds__(128)
void gemm_wmma_k(const bf16_t* __restrict__ A, const bf16_t* __restrict__ W,
                 const float* __restrict__ bias, float* __restrict__ Cf,
                 bf16_t* __restrict__ Cb, int M, int N, int K,
                 float scale, int relu, int transOut, int ldOut) {
  __shared__ __align__(16) bf16_t Wlds[2][64][32];

  int tid  = threadIdx.x;
  int lane = tid & 31;
  int wid  = tid >> 5;
  int tnb  = blockIdx.x << 6;                 // block N base (64 cols)
  int tm   = (blockIdx.y * 4 + wid) << 4;     // wave M tile
  int l15  = lane & 15;
  int hi   = lane >> 4;
  int kbA  = hi << 4;    // A frag: halves {kbA..kbA+15}
  int kbB  = hi << 4;    // B frag: halves {kbB..kbB+15}

  int arow = tm + l15;
  if (arow >= M) arow = M - 1;                // keep OOB tiles in-bounds
  const bf16_t* Ar = A + (size_t)arow * K;

  // stage W[tnb..tnb+63][k0..k0+31] -> Wlds[buf] (4KB, 2x b128 per thread)
  auto stage = [&](int buf, int k0) {
#pragma unroll
    for (int it = 0; it < 2; ++it) {
      int t   = tid + it * 128;
      int c   = t >> 2;
      int seg = (t & 3) << 3;
      const bf16_t* g = W + (size_t)(tnb + c) * K + k0 + seg;
      bf16_t* l = &Wlds[buf][c][seg];
#if HAVE_ASYNC
      __builtin_amdgcn_global_load_async_to_lds_b128(
          (AS_GLOBAL v4i_vec*)g, (AS_LDS v4i_vec*)l, 0, 0);
#else
      *(uint4*)l = *(const uint4*)g;
#endif
    }
  };

  v8f acc[4];
#pragma unroll
  for (int nt = 0; nt < 4; ++nt) acc[nt] = {0.f,0.f,0.f,0.f,0.f,0.f,0.f,0.f};

  stage(0, 0);
  int buf = 0;
  for (int k0 = 0; k0 < K; k0 += 32) {
    bool more = (k0 + 32) < K;
    if (more) stage(buf ^ 1, k0 + 32);        // overlap next W tile fetch
#if HAVE_ASYNC
    if (more) __builtin_amdgcn_s_wait_asynccnt(2);  // current tile landed
    else      __builtin_amdgcn_s_wait_asynccnt(0);
#endif
    __syncthreads();

    union { v16bf v; uint4 u[2]; } a;
    a.u[0] = *(const uint4*)(Ar + k0 + kbA);
    a.u[1] = *(const uint4*)(Ar + k0 + kbA + 8);
    if (more) __builtin_prefetch(Ar + k0 + 32, 0, 3);   // global_prefetch_b8

#pragma unroll
    for (int nt = 0; nt < 4; ++nt) {
      union { v16bf v; uint4 u[2]; } b;
      const bf16_t* lp = &Wlds[buf][(nt << 4) + l15][kbB];
      b.u[0] = *(const uint4*)lp;
      b.u[1] = *(const uint4*)(lp + 8);
      acc[nt] = __builtin_amdgcn_wmma_f32_16x16x32_bf16(false, a.v, false, b.v,
                                                        (short)0, acc[nt], false, false);
    }
    __syncthreads();   // all waves done with buf before it is restaged
    buf ^= 1;
  }

#pragma unroll
  for (int nt = 0; nt < 4; ++nt) {
    int col = tnb + (nt << 4) + l15;
    float bv = bias ? bias[col] : 0.f;
#pragma unroll
    for (int j = 0; j < 8; ++j) {
      int r = tm + j + (hi << 3);
      if (r >= M) continue;
      float v = (acc[nt][j] + bv) * scale;
      if (relu) v = fmaxf(v, 0.f);
      if (Cf) Cf[(size_t)r * N + col] = v;
      if (Cb) {
        if (transOut) Cb[(size_t)col * ldOut + r] = (bf16_t)v;
        else          Cb[(size_t)r * N + col]     = (bf16_t)v;
      }
    }
  }
}

// ---------------------------------------------------------------------------
// Flash attention: one wave per (16-query tile, head).
// Qh/Kh: [rows][D] bf16, head h in cols h*32..h*32+31 (Q pre-scaled).
// Vt:    [D][skp] bf16 (transposed V).  mask: uint8 [NQ][maskStride], 1=block.
// O:     [NQP][D] bf16.
// ---------------------------------------------------------------------------
__global__ __launch_bounds__(32)
void attn_wmma_k(const bf16_t* __restrict__ Qh, const bf16_t* __restrict__ Kh,
                 const bf16_t* __restrict__ Vt, const unsigned char* __restrict__ mask,
                 int maskStride, bf16_t* __restrict__ O, int skp, int sk) {
  __shared__ float  Sts[16][33];
  __shared__ __align__(16) bf16_t Pb[16][32];
  __shared__ float  mrun[16], lrun[16], resc[16];
  const float NI = neg_inf();

  int lane  = threadIdx.x;
  int qtile = blockIdx.x << 4;
  int h     = blockIdx.y;
  int l15   = lane & 15;
  int hi    = lane >> 4;
  int kbA   = hi << 4;
  int kbB   = hi << 4;

  union { v16bf v; uint4 u[2]; } qa;
  const bf16_t* Qr = Qh + (size_t)(qtile + l15) * D_MODEL + h * HDIM;
  qa.u[0] = *(const uint4*)(Qr + kbA);
  qa.u[1] = *(const uint4*)(Qr + kbA + 8);

  v8f o0 = {0.f,0.f,0.f,0.f,0.f,0.f,0.f,0.f};
  v8f o1 = {0.f,0.f,0.f,0.f,0.f,0.f,0.f,0.f};
  if (lane < 16) { mrun[lane] = NI; lrun[lane] = 0.f; }
  __syncthreads();

  const bf16_t* V0base = Vt + (size_t)(h * HDIM + l15)      * skp;
  const bf16_t* V1base = Vt + (size_t)(h * HDIM + 16 + l15) * skp;

  for (int kc = 0; kc < sk; kc += 32) {
    union { v16bf v; uint4 u[2]; } b0, b1, pa, vb;
    int key0 = kc + l15;
    int key1 = kc + 16 + l15;
    if (key0 < sk) {
      const bf16_t* Kr = Kh + (size_t)key0 * D_MODEL + h * HDIM + kbB;
      b0.u[0] = *(const uint4*)Kr; b0.u[1] = *(const uint4*)(Kr + 8);
    } else { b0.u[0] = make_uint4(0,0,0,0); b0.u[1] = make_uint4(0,0,0,0); }
    if (key1 < sk) {
      const bf16_t* Kr = Kh + (size_t)key1 * D_MODEL + h * HDIM + kbB;
      b1.u[0] = *(const uint4*)Kr; b1.u[1] = *(const uint4*)(Kr + 8);
    } else { b1.u[0] = make_uint4(0,0,0,0); b1.u[1] = make_uint4(0,0,0,0); }

    v8f z = {0.f,0.f,0.f,0.f,0.f,0.f,0.f,0.f};
    v8f s0 = __builtin_amdgcn_wmma_f32_16x16x32_bf16(false, qa.v, false, b0.v,
                                                     (short)0, z, false, false);
    v8f s1 = __builtin_amdgcn_wmma_f32_16x16x32_bf16(false, qa.v, false, b1.v,
                                                     (short)0, z, false, false);
#pragma unroll
    for (int j = 0; j < 8; ++j) {
      int m = j + (hi << 3);
      int q = qtile + m;
      bool ok0 = (kc + l15) < sk;
      bool ok1 = (kc + 16 + l15) < sk;
      if (mask != nullptr && q < NQ) {
        const unsigned char* mr = mask + (size_t)q * maskStride;
        if (ok0 && mr[kc + l15])      ok0 = false;
        if (ok1 && mr[kc + 16 + l15]) ok1 = false;
      }
      Sts[m][l15]      = ok0 ? s0[j] : NI;
      Sts[m][16 + l15] = ok1 ? s1[j] : NI;
    }
    __syncthreads();

    if (lane < 16) {  // online softmax for row `lane`
      int r = lane;
      float mold = mrun[r];
      float mx = mold;
#pragma unroll
      for (int c = 0; c < 32; ++c) mx = fmaxf(mx, Sts[r][c]);
      float corr = (mx == NI) ? 1.f : __expf(mold - mx);
      float sum = lrun[r] * corr;
#pragma unroll
      for (int c = 0; c < 32; ++c) {
        float sv = Sts[r][c];
        float p = (sv == NI) ? 0.f : __expf(sv - mx);
        Pb[r][c] = (bf16_t)p;
        sum += p;
      }
      mrun[r] = mx; lrun[r] = sum; resc[r] = corr;
    }
    __syncthreads();

#pragma unroll
    for (int j = 0; j < 8; ++j) {
      float c = resc[j + (hi << 3)];
      o0[j] *= c; o1[j] *= c;
    }
    pa.u[0] = *(const uint4*)(&Pb[l15][kbA]);
    pa.u[1] = *(const uint4*)(&Pb[l15][kbA + 8]);

    vb.u[0] = *(const uint4*)(V0base + kc + kbB);
    vb.u[1] = *(const uint4*)(V0base + kc + kbB + 8);
    o0 = __builtin_amdgcn_wmma_f32_16x16x32_bf16(false, pa.v, false, vb.v,
                                                 (short)0, o0, false, false);
    vb.u[0] = *(const uint4*)(V1base + kc + kbB);
    vb.u[1] = *(const uint4*)(V1base + kc + kbB + 8);
    o1 = __builtin_amdgcn_wmma_f32_16x16x32_bf16(false, pa.v, false, vb.v,
                                                 (short)0, o1, false, false);
    __syncthreads();
  }

#pragma unroll
  for (int j = 0; j < 8; ++j) {
    int m = j + (hi << 3);
    float l = lrun[m];
    float inv = (l > 0.f) ? 1.f / l : 0.f;
    bf16_t* Or = O + (size_t)(qtile + m) * D_MODEL + h * HDIM;
    Or[l15]      = (bf16_t)(o0[j] * inv);
    Or[16 + l15] = (bf16_t)(o1[j] * inv);
  }
}

// ---------------------------------------------------------------------------
// Residual + LayerNorm (one wave per row). Writes fp32 tgt, bf16 tgt,
// and bf16(tgt + qpos) for the next attention's q/k input.
// ---------------------------------------------------------------------------
__global__ __launch_bounds__(32)
void ln_residual_k(const float* __restrict__ x, const float* __restrict__ hres,
                   const float* __restrict__ g, const float* __restrict__ b,
                   const float* __restrict__ qpos, float* __restrict__ xout,
                   bf16_t* __restrict__ xbf, bf16_t* __restrict__ qbf) {
  int row = blockIdx.x;
  int lane = threadIdx.x;
  float v[8];
  float s = 0.f;
#pragma unroll
  for (int i = 0; i < 8; ++i) {
    int c = lane + (i << 5);
    v[i] = x[(size_t)row * D_MODEL + c] + hres[(size_t)row * D_MODEL + c];
    s += v[i];
  }
#pragma unroll
  for (int off = 16; off >= 1; off >>= 1) s += __shfl_xor(s, off, 32);
  float mu = s * (1.f / 256.f);
  float vs = 0.f;
#pragma unroll
  for (int i = 0; i < 8; ++i) { float d = v[i] - mu; vs += d * d; }
#pragma unroll
  for (int off = 16; off >= 1; off >>= 1) vs += __shfl_xor(vs, off, 32);
  float rstd = rsqrtf(vs * (1.f / 256.f) + 1e-5f);
#pragma unroll
  for (int i = 0; i < 8; ++i) {
    int c = lane + (i << 5);
    float y = (v[i] - mu) * rstd * g[c] + b[c];
    xout[(size_t)row * D_MODEL + c] = y;
    xbf[(size_t)row * D_MODEL + c] = (bf16_t)y;
    qbf[(size_t)row * D_MODEL + c] = (bf16_t)(y + qpos[(size_t)row * D_MODEL + c]);
  }
}

// ---------------------------------------------------------------------------
// host launcher
// ---------------------------------------------------------------------------
extern "C" void kernel_launch(void* const* d_in, const int* in_sizes, int n_in,
                              void* d_out, int out_size, void* d_ws, size_t ws_size,
                              hipStream_t stream) {
  (void)in_sizes; (void)n_in; (void)out_size; (void)ws_size;
  const float* src  = (const float*)d_in[0];
  const float* pos  = (const float*)d_in[1];
  const float* qe   = (const float*)d_in[2];
  const float* Wqkv_s = (const float*)d_in[3];
  const float* bqkv_s = (const float*)d_in[4];
  const float* bo_s   = (const float*)d_in[6];
  const float* Wqkv_o = (const float*)d_in[7];
  const float* bqkv_o = (const float*)d_in[8];
  const float* bo_o   = (const float*)d_in[10];
  const float* Wqkv_c = (const float*)d_in[11];
  const float* bqkv_c = (const float*)d_in[12];
  const float* bo_c   = (const float*)d_in[14];
  const float* W1     = (const float*)d_in[15];
  const float* b1     = (const float*)d_in[16];
  const float* W2     = (const float*)d_in[17];
  const float* b2     = (const float*)d_in[18];
  const float* g1 = (const float*)d_in[19];
  const float* n1 = (const float*)d_in[20];
  const float* g2 = (const float*)d_in[21];
  const float* n2 = (const float*)d_in[22];
  const float* go = (const float*)d_in[23];
  const float* no = (const float*)d_in[24];
  const float* g3 = (const float*)d_in[25];
  const float* n3 = (const float*)d_in[26];
  const unsigned char* attn_mask = (const unsigned char*)d_in[27];
  const unsigned char* obj_mask  = (const unsigned char*)d_in[28];

  char* wp = (char*)d_ws;
  auto alloc = [&](size_t bytes) -> char* {
    char* p = wp; wp += (bytes + 255) & ~(size_t)255; return p;
  };
  const size_t DD = (size_t)D_MODEL * D_MODEL;
  bf16_t* wqkv_s_bf = (bf16_t*)alloc(LAY * 3 * DD * 2);
  bf16_t* wqkv_o_bf = (bf16_t*)alloc(LAY * 3 * DD * 2);
  bf16_t* wqkv_c_bf = (bf16_t*)alloc(LAY * 3 * DD * 2);
  bf16_t* wo_s_bf   = (bf16_t*)alloc(LAY * DD * 2);
  bf16_t* wo_o_bf   = (bf16_t*)alloc(LAY * DD * 2);
  bf16_t* wo_c_bf   = (bf16_t*)alloc(LAY * DD * 2);
  bf16_t* w1_bf     = (bf16_t*)alloc((size_t)LAY * DFF * D_MODEL * 2);
  bf16_t* w2_bf     = (bf16_t*)alloc((size_t)LAY * D_MODEL * DFF * 2);
  bf16_t* s_bf      = (bf16_t*)alloc((size_t)SRC_N * D_MODEL * 2);
  bf16_t* kp_bf     = (bf16_t*)alloc((size_t)SRC_N * D_MODEL * 2);
  float*  tgt       = (float*) alloc((size_t)NQP * D_MODEL * 4);
  float*  qpos      = (float*) alloc((size_t)NQP * D_MODEL * 4);
  float*  proj      = (float*) alloc((size_t)NQP * D_MODEL * 4);
  bf16_t* tgt_bf    = (bf16_t*)alloc((size_t)NQP * D_MODEL * 2);
  bf16_t* q_bf      = (bf16_t*)alloc((size_t)NQP * D_MODEL * 2);
  bf16_t* qh_bf     = (bf16_t*)alloc((size_t)NQP * D_MODEL * 2);
  bf16_t* kh_self   = (bf16_t*)alloc((size_t)NQP * D_MODEL * 2);
  bf16_t* vt_self   = (bf16_t*)alloc((size_t)D_MODEL * NQP * 2);
  bf16_t* kh_src    = (bf16_t*)alloc((size_t)SRC_N * D_MODEL * 2);
  bf16_t* vt_src    = (bf16_t*)alloc((size_t)D_MODEL * SRC_N * 2);
  bf16_t* attn_bf   = (bf16_t*)alloc((size_t)NQP * D_MODEL * 2);
  bf16_t* h_bf      = (bf16_t*)alloc((size_t)NQP * DFF * 2);

  auto conv = [&](const float* s, bf16_t* dst, size_t n) {
    f32_to_bf16_k<<<dim3((unsigned)((n + 255) / 256)), 256, 0, stream>>>(s, dst, (int)n);
  };
  conv(Wqkv_s, wqkv_s_bf, LAY * 3 * DD);
  conv(Wqkv_o, wqkv_o_bf, LAY * 3 * DD);
  conv(Wqkv_c, wqkv_c_bf, LAY * 3 * DD);
  conv((const float*)d_in[5],  wo_s_bf, LAY * DD);
  conv((const float*)d_in[9],  wo_o_bf, LAY * DD);
  conv((const float*)d_in[13], wo_c_bf, LAY * DD);
  conv(W1, w1_bf, (size_t)LAY * DFF * D_MODEL);
  conv(W2, w2_bf, (size_t)LAY * D_MODEL * DFF);

  prep_src_k<<<dim3((SRC_N * D_MODEL + 255) / 256), 256, 0, stream>>>(src, pos, s_bf, kp_bf);
  prep_q_k<<<dim3((NQP * D_MODEL + 255) / 256), 256, 0, stream>>>(qe, tgt, qpos, tgt_bf, q_bf);

  const float QSC = 0.17677669529663687f;  // 1/sqrt(32)
  auto gemm = [&](const bf16_t* A, const bf16_t* W, const float* bias,
                  float* Cf, bf16_t* Cb, int M, int N, int K,
                  float scale, int relu, int trans, int ldo) {
    dim3 grid(N / 64, (M + 63) / 64);
    gemm_wmma_k<<<grid, 128, 0, stream>>>(A, W, bias, Cf, Cb, M, N, K, scale, relu, trans, ldo);
  };
  auto attn = [&](const bf16_t* Q, const bf16_t* K, const bf16_t* Vt,
                  const unsigned char* mask, int mstride, bf16_t* O, int skp, int sk) {
    attn_wmma_k<<<dim3(NQP / 16, NHEAD), 32, 0, stream>>>(Q, K, Vt, mask, mstride, O, skp, sk);
  };
  auto ln = [&](const float* g, const float* b) {
    ln_residual_k<<<dim3(NQP), 32, 0, stream>>>(tgt, proj, g, b, qpos, tgt, tgt_bf, q_bf);
  };

  for (int l = 0; l < LAY; ++l) {
    // ---- plain self-attention -------------------------------------------
    const bf16_t* Wq = wqkv_s_bf + (size_t)l * 3 * DD;
    const float*  bq = bqkv_s + (size_t)l * 3 * D_MODEL;
    gemm(q_bf,   Wq,           bq,               nullptr, qh_bf,  NQP, D_MODEL, D_MODEL, QSC, 0, 0, 0);
    gemm(q_bf,   Wq + DD,      bq + D_MODEL,     nullptr, kh_self,NQP, D_MODEL, D_MODEL, 1.f, 0, 0, 0);
    gemm(tgt_bf, Wq + 2 * DD,  bq + 2 * D_MODEL, nullptr, vt_self,NQP, D_MODEL, D_MODEL, 1.f, 0, 1, NQP);
    attn(qh_bf, kh_self, vt_self, nullptr, 0, attn_bf, NQP, NQ);
    gemm(attn_bf, wo_s_bf + (size_t)l * DD, bo_s + (size_t)l * D_MODEL,
         proj, nullptr, NQP, D_MODEL, D_MODEL, 1.f, 0, 0, 0);
    ln(g2 + (size_t)l * D_MODEL, n2 + (size_t)l * D_MODEL);

    // ---- block-masked object self-attention -----------------------------
    Wq = wqkv_o_bf + (size_t)l * 3 * DD;
    bq = bqkv_o + (size_t)l * 3 * D_MODEL;
    gemm(q_bf,   Wq,           bq,               nullptr, qh_bf,  NQP, D_MODEL, D_MODEL, QSC, 0, 0, 0);
    gemm(q_bf,   Wq + DD,      bq + D_MODEL,     nullptr, kh_self,NQP, D_MODEL, D_MODEL, 1.f, 0, 0, 0);
    gemm(tgt_bf, Wq + 2 * DD,  bq + 2 * D_MODEL, nullptr, vt_self,NQP, D_MODEL, D_MODEL, 1.f, 0, 1, NQP);
    attn(qh_bf, kh_self, vt_self, obj_mask, NQ, attn_bf, NQP, NQ);
    gemm(attn_bf, wo_o_bf + (size_t)l * DD, bo_o + (size_t)l * D_MODEL,
         proj, nullptr, NQP, D_MODEL, D_MODEL, 1.f, 0, 0, 0);
    ln(go + (size_t)l * D_MODEL, no + (size_t)l * D_MODEL);

    // ---- bbox-masked cross-attention ------------------------------------
    Wq = wqkv_c_bf + (size_t)l * 3 * DD;
    bq = bqkv_c + (size_t)l * 3 * D_MODEL;
    gemm(q_bf,  Wq,          bq,               nullptr, qh_bf,  NQP,   D_MODEL, D_MODEL, QSC, 0, 0, 0);
    gemm(kp_bf, Wq + DD,     bq + D_MODEL,     nullptr, kh_src, SRC_N, D_MODEL, D_MODEL, 1.f, 0, 0, 0);
    gemm(s_bf,  Wq + 2 * DD, bq + 2 * D_MODEL, nullptr, vt_src, SRC_N, D_MODEL, D_MODEL, 1.f, 0, 1, SRC_N);
    attn(qh_bf, kh_src, vt_src, attn_mask, SRC_N, attn_bf, SRC_N, SRC_N);
    gemm(attn_bf, wo_c_bf + (size_t)l * DD, bo_c + (size_t)l * D_MODEL,
         proj, nullptr, NQP, D_MODEL, D_MODEL, 1.f, 0, 0, 0);
    ln(g1 + (size_t)l * D_MODEL, n1 + (size_t)l * D_MODEL);

    // ---- FFN -------------------------------------------------------------
    gemm(tgt_bf, w1_bf + (size_t)l * DFF * D_MODEL, b1 + (size_t)l * DFF,
         nullptr, h_bf, NQP, DFF, D_MODEL, 1.f, 1, 0, 0);
    gemm(h_bf, w2_bf + (size_t)l * D_MODEL * DFF, b2 + (size_t)l * D_MODEL,
         proj, nullptr, NQP, D_MODEL, DFF, 1.f, 0, 0, 0);
    ln(g3 + (size_t)l * D_MODEL, n3 + (size_t)l * D_MODEL);
  }

  copy_out_k<<<dim3((NQ * D_MODEL + 255) / 256), 256, 0, stream>>>(tgt, (float*)d_out, NQ * D_MODEL);
}
